// MoELayer_51247549776482
// MI455X (gfx1250) — matrix operand order repaired and checked by reading, not compiled, not run on hardware
//
#include <hip/hip_runtime.h>
#include <hip/hip_bf16.h>
#include <cstddef>

// ---------------------------------------------------------------------------
// MoE layer (top-2 of 8 experts + shared expert) for gfx1250 (MI455X).
// HBM-bandwidth bound (~340MB f32 @ 23.3 TB/s ~ 15us): sparse per-expert
// token gather (4x fewer FLOPs than dense dispatch), f32->bf16 conversion in
// LDS, v_wmma_f32_16x16x32_bf16 for the math. LDS tiles are laid out so each
// lane's WMMA fragment is two aligned 16B runs -> 6x ds_load_b128 feed 2 WMMAs
// per K-step. Staging uses invariant base pointers + scalar offsets so loads
// lower to global_load (not flat_load), and no predication in the hot loop:
// invalid gather rows/cols are clamped in-bounds and masked at the epilogue.
// ---------------------------------------------------------------------------

typedef __attribute__((ext_vector_type(16))) __bf16 v16bf;
typedef __attribute__((ext_vector_type(8)))  __bf16 v8bf;
typedef __attribute__((ext_vector_type(8)))  float  v8f;

#define TK_TOKENS 4096   // B*T
#define CDIM      1024
#define EXP       8
#define TOPK      2
#define HDIM      2736
#define HMAIN     2720   // 85 full 32-wide K slices; 16-wide guarded tail
#define HROWS     12288  // 8192 routed (t*2+k) + 4096 shared (8192+t)

// bf16 round-to-nearest-even
static __device__ __forceinline__ unsigned short f32_to_bf16_bits(float f) {
    unsigned u = __builtin_bit_cast(unsigned, f);
    unsigned r = u + 0x7FFFu + ((u >> 16) & 1u);
    return (unsigned short)(r >> 16);
}
static __device__ __forceinline__ unsigned pack2_bf16(float a, float b) {
    return (unsigned)f32_to_bf16_bits(a) | ((unsigned)f32_to_bf16_bits(b) << 16);
}
static __device__ __forceinline__ v16bf cat8(v8bf lo, v8bf hi) {
    return __builtin_shufflevector(lo, hi, 0, 1, 2, 3, 4, 5, 6, 7,
                                           8, 9, 10, 11, 12, 13, 14, 15);
}
// A fragment: row-major [M=32][K=32] bf16 tile; two aligned 16B runs per lane
static __device__ __forceinline__ v16bf frag_A(const __bf16* As, int lane, int wm) {
    const v8bf* Ap = (const v8bf*)As;
    int r = (lane & 15) + wm * 16;
    int base = r * 4 + ((lane < 16) ? 0 : 2);   // K base 0 or 16 (/8)
    return cat8(Ap[base], Ap[base + 1]);
}
// B fragment: transposed [N=64][K=32] bf16 tile; one 32B run per lane
static __device__ __forceinline__ v16bf frag_B(const __bf16* Bs, int lane, int wn) {
    const v8bf* Bp = (const v8bf*)Bs;
    int cN = wn * 16 + (lane & 15);
    int base = cN * 4 + ((lane < 16) ? 0 : 2);  // K base 0 or 16 (/8)
    return cat8(Bp[base], Bp[base + 1]);
}
// fast SwiGLU activation: x * sigmoid(x) via v_rcp_f32 (bf16-precision pipe,
// exact-rounded IEEE divide expansion is wasted VALU work here)
static __device__ __forceinline__ float silu_fast(float g) {
    return g * __builtin_amdgcn_rcpf(1.0f + __expf(-g));
}

// ---------------------------------------------------------------------------
// zero-init kernels
// ---------------------------------------------------------------------------
__global__ void moe_zero_f32(float* p, size_t n) {
    size_t i = (size_t)blockIdx.x * blockDim.x + threadIdx.x;
    size_t stride = (size_t)gridDim.x * blockDim.x;
    for (; i < n; i += stride) p[i] = 0.0f;
}

__global__ void moe_zero_ctr(unsigned* fill, float* psum) {
    int t = threadIdx.x;
    if (t < 16) fill[t] = 0u;
    if (t < EXP) psum[t] = 0.0f;
}

// ---------------------------------------------------------------------------
// Router: one wave32 per token; wave-reduced logits, top-2 softmax, atomic
// per-expert gather lists, softmax sums for the aux loss.
// ---------------------------------------------------------------------------
__global__ void moe_router(const float* __restrict__ x,
                           const float* __restrict__ rw,
                           unsigned* __restrict__ fill,
                           float* __restrict__ psum,
                           unsigned* __restrict__ list,
                           float* __restrict__ hw) {
    int lane = threadIdx.x & 31;
    int wv   = threadIdx.x >> 5;
    int t = blockIdx.x * 8 + wv;
    if (t >= TK_TOKENS) return;

    float acc[EXP];
#pragma unroll
    for (int e = 0; e < EXP; ++e) acc[e] = 0.0f;

    const float* xr = x + (size_t)t * CDIM;
    for (int c = lane; c < CDIM; c += 32) {
        float xv = xr[c];
        const float* rr = rw + (size_t)c * EXP;
#pragma unroll
        for (int e = 0; e < EXP; ++e) acc[e] += xv * rr[e];
    }
#pragma unroll
    for (int e = 0; e < EXP; ++e) {
        for (int off = 16; off > 0; off >>= 1)
            acc[e] += __shfl_xor(acc[e], off, 32);
    }

    if (lane == 0) {
        int i0 = 0;
#pragma unroll
        for (int e = 1; e < EXP; ++e) if (acc[e] > acc[i0]) i0 = e;
        int i1 = (i0 == 0) ? 1 : 0;
#pragma unroll
        for (int e = 0; e < EXP; ++e)
            if (e != i0 && acc[e] > acc[i1]) i1 = e;

        float v0 = acc[i0], v1 = acc[i1];
        float mx = fmaxf(v0, v1);
        float e0 = __expf(v0 - mx), e1 = __expf(v1 - mx);
        float inv = 1.0f / (e0 + e1);
        float w0 = e0 * inv, w1 = e1 * inv;

        float fm = acc[0];
#pragma unroll
        for (int e = 1; e < EXP; ++e) fm = fmaxf(fm, acc[e]);
        float pex[EXP], s = 0.0f;
#pragma unroll
        for (int e = 0; e < EXP; ++e) { pex[e] = __expf(acc[e] - fm); s += pex[e]; }
        float is = 1.0f / s;
#pragma unroll
        for (int e = 0; e < EXP; ++e) atomicAdd(&psum[e], pex[e] * is);

        unsigned p0 = atomicAdd(&fill[i0], 1u);
        list[(size_t)i0 * TK_TOKENS + p0] = (unsigned)(t * 2 + 0);
        hw[t * 2 + 0] = w0;
        unsigned p1 = atomicAdd(&fill[i1], 1u);
        list[(size_t)i1 * TK_TOKENS + p1] = (unsigned)(t * 2 + 1);
        hw[t * 2 + 1] = w1;
    }
}

// aux_loss = E * sum_e (count_e/(Tk*K)) * (psum_e/Tk)
__global__ void moe_aux(const unsigned* __restrict__ fill,
                        const float* __restrict__ psum,
                        float* __restrict__ out) {
    if (threadIdx.x == 0 && blockIdx.x == 0) {
        float s = 0.0f;
        for (int e = 0; e < EXP; ++e) {
            float f = (float)fill[e] / (float)(TK_TOKENS * TOPK);
            float p = psum[e] / (float)TK_TOKENS;
            s += f * p;
        }
        out[(size_t)TK_TOKENS * CDIM] = (float)EXP * s;
    }
}

// ---------------------------------------------------------------------------
// Gather-list metadata for a 32-row tile.
// ---------------------------------------------------------------------------
static __device__ __forceinline__ void load_rows(int e, int m0, int cnt,
                                                 const unsigned* list,
                                                 const float* hw,
                                                 int* s_tok, unsigned* s_hr,
                                                 float* s_w) {
    int tid = threadIdx.x;
    if (tid < 32) {
        int i = m0 + tid;
        if (i < cnt) {
            if (e < EXP) {
                unsigned v = list[(size_t)e * TK_TOKENS + i];
                s_tok[tid] = (int)(v >> 1);
                s_hr[tid]  = v;                    // routed rows 0..8191
                if (s_w) s_w[tid] = hw[v];
            } else {
                s_tok[tid] = i;
                s_hr[tid]  = 8192u + (unsigned)i;  // shared-expert rows
                if (s_w) s_w[tid] = 1.0f;
            }
        } else { s_tok[tid] = -1; s_hr[tid] = 0u; if (s_w) s_w[tid] = 0.0f; }
    }
}

// ---------------------------------------------------------------------------
// Fused gate+up GEMM + SwiGLU + routing-weight scale.
// grid = (ceil(H/64), Tk/32, 9); z = expert slot, slot 8 = shared expert.
// 256 threads = 8 waves; 32x64 block tile; per wave one 16x16 WMMA tile with
// two accumulators (gate, up) sharing the A fragment.
// ---------------------------------------------------------------------------
__global__ void moe_gateup(const float* __restrict__ x,
                           const float* __restrict__ wg,
                           const float* __restrict__ wu,
                           const float* __restrict__ sg,
                           const float* __restrict__ su,
                           const unsigned* __restrict__ fill,
                           const unsigned* __restrict__ list,
                           const float* __restrict__ hw,
                           unsigned short* __restrict__ hbuf) {
    constexpr int TM = 32, TN = 64, TKK = 32;
    __shared__ __attribute__((aligned(16))) __bf16 As[TM * TKK];  // [m][k]
    __shared__ __attribute__((aligned(16))) __bf16 Bg[TN * TKK];  // [n][k]
    __shared__ __attribute__((aligned(16))) __bf16 Bu[TN * TKK];  // [n][k]
    __shared__ int      s_tok[TM];
    __shared__ unsigned s_hr[TM];
    __shared__ float    s_w[TM];

    int e  = blockIdx.z;
    int m0 = blockIdx.y * TM;
    int n0 = blockIdx.x * TN;

    const float* gw;
    const float* uw;
    int cnt;
    if (e < EXP) {
        cnt = (int)fill[e];
        gw = wg + (size_t)e * CDIM * HDIM;
        uw = wu + (size_t)e * CDIM * HDIM;
    } else {
        cnt = TK_TOKENS;
        gw = sg;
        uw = su;
    }
    if (m0 >= cnt) return;   // uniform per block

    int tid = threadIdx.x;
    load_rows(e, m0, cnt, list, hw, s_tok, s_hr, s_w);
    __syncthreads();

    int lane = tid & 31;
    int wv   = tid >> 5;
    int wm   = wv >> 2;   // 0..1 -> M offset 16*wm
    int wn   = wv & 3;    // 0..3 -> N offset 16*wn

    unsigned* Aw = (unsigned*)As;  // u32 = 2 packed bf16 along K
    unsigned* Gw = (unsigned*)Bg;
    unsigned* Uw = (unsigned*)Bu;

    // ---- hoisted staging offsets (invalid rows/cols clamped in-bounds;
    //      their results are masked at the epilogue store) ----
    size_t aoff[2]; int aloc[2];
#pragma unroll
    for (int it = 0; it < 2; ++it) {
        int idx = tid + 256 * it;
        int r = idx >> 4, cp = idx & 15;
        int tok = s_tok[r];
        int tr  = tok < 0 ? 0 : tok;
        aoff[it] = (size_t)tr * CDIM + 2 * cp;
        aloc[it] = r * 16 + cp;
    }
    size_t boff[4]; int bloc[4];
#pragma unroll
    for (int it = 0; it < 4; ++it) {
        int idx = tid + 256 * it;
        int kp = idx >> 6, n = idx & 63;     // consecutive tid -> consecutive n
        int gn  = n0 + n;
        int gnc = gn >= HDIM ? (HDIM - 1) : gn;
        boff[it] = (size_t)(2 * kp) * HDIM + gnc;
        bloc[it] = n * 16 + kp;
    }

    v8f cg = {};
    v8f cu = {};

    size_t kofs = 0;
    for (int k0 = 0; k0 < CDIM; k0 += TKK) {
#pragma unroll
        for (int it = 0; it < 2; ++it) {
            float2 v = *(const float2*)(x + aoff[it] + k0);
            Aw[aloc[it]] = pack2_bf16(v.x, v.y);
        }
#pragma unroll
        for (int it = 0; it < 4; ++it) {
            size_t o = boff[it] + kofs;
            float g0 = gw[o], g1 = gw[o + HDIM];
            float u0 = uw[o], u1 = uw[o + HDIM];
            Gw[bloc[it]] = pack2_bf16(g0, g1);
            Uw[bloc[it]] = pack2_bf16(u0, u1);
        }
        kofs += (size_t)TKK * HDIM;
        __syncthreads();

        v16bf a  = frag_A(As, lane, wm);
        v16bf bg = frag_B(Bg, lane, wn);
        v16bf bu = frag_B(Bu, lane, wn);
        cg = __builtin_amdgcn_wmma_f32_16x16x32_bf16(false, a, false, bg,
                                                     (short)0, cg, false, false);
        cu = __builtin_amdgcn_wmma_f32_16x16x32_bf16(false, a, false, bu,
                                                     (short)0, cu, false, false);
        __syncthreads();
    }

    // epilogue: SwiGLU, routing-weight scale, bf16 store of h
    int mb = wm * 16 + ((lane < 16) ? 0 : 8);
    int nn = n0 + wn * 16 + (lane & 15);
#pragma unroll
    for (int v = 0; v < 8; ++v) {
        int m = mb + v;
        int tok = s_tok[m];
        if (tok >= 0 && nn < HDIM) {
            float hval = silu_fast(cg[v]) * cu[v] * s_w[m];
            hbuf[(size_t)s_hr[m] * HDIM + nn] = f32_to_bf16_bits(hval);
        }
    }
}

// ---------------------------------------------------------------------------
// Down projection: h (gathered bf16 rows) @ wd_e -> atomicAdd into out.
// grid = (C/64, Tk/32, 9). 85 full K slices + one guarded 16-wide tail
// (K contributes to every output element, so tail padding is mandatory).
// ---------------------------------------------------------------------------
__global__ void moe_down(const unsigned short* __restrict__ hbuf,
                         const float* __restrict__ wd,
                         const float* __restrict__ sd,
                         const unsigned* __restrict__ fill,
                         const unsigned* __restrict__ list,
                         float* __restrict__ out) {
    constexpr int TM = 32, TN = 64, TKK = 32;
    __shared__ __attribute__((aligned(16))) __bf16 As[TM * TKK];  // [m][k]
    __shared__ __attribute__((aligned(16))) __bf16 Bs[TN * TKK];  // [n][k]
    __shared__ int      s_tok[TM];
    __shared__ unsigned s_hr[TM];

    int e  = blockIdx.z;
    int m0 = blockIdx.y * TM;
    int n0 = blockIdx.x * TN;

    const float* dwp;
    int cnt;
    if (e < EXP) {
        cnt = (int)fill[e];
        dwp = wd + (size_t)e * HDIM * CDIM;
    } else {
        cnt = TK_TOKENS;
        dwp = sd;
    }
    if (m0 >= cnt) return;

    int tid = threadIdx.x;
    load_rows(e, m0, cnt, list, nullptr, s_tok, s_hr, nullptr);
    __syncthreads();

    int lane = tid & 31;
    int wv   = tid >> 5;
    int wm   = wv >> 2;
    int wn   = wv & 3;

    unsigned* Aw = (unsigned*)As;
    unsigned* Bw = (unsigned*)Bs;
    const unsigned* hbw = (const unsigned*)hbuf;   // u32 view (HDIM even)

    // ---- hoisted staging offsets ----
    size_t aoff[2]; int aloc[2]; int acp[2];
#pragma unroll
    for (int it = 0; it < 2; ++it) {
        int idx = tid + 256 * it;
        int r = idx >> 4, cp = idx & 15;
        acp[it]  = cp;
        aoff[it] = (size_t)s_hr[r] * (HDIM / 2) + cp;  // u32 units
        aloc[it] = r * 16 + cp;
    }
    size_t boff[4]; int bloc[4]; int bkp[4];
#pragma unroll
    for (int it = 0; it < 4; ++it) {
        int idx = tid + 256 * it;
        int kp = idx >> 6, n = idx & 63;
        bkp[it]  = kp;
        boff[it] = (size_t)(2 * kp) * CDIM + (n0 + n);
        bloc[it] = n * 16 + kp;
    }

    v8f cacc = {};

    // main loop: no K guards (k0+31 < HMAIN <= HDIM)
    size_t ka = 0, kb = 0;
    for (int k0 = 0; k0 < HMAIN; k0 += TKK) {
#pragma unroll
        for (int it = 0; it < 2; ++it)
            Aw[aloc[it]] = hbw[aoff[it] + ka];
#pragma unroll
        for (int it = 0; it < 4; ++it) {
            size_t o = boff[it] + kb;
            Bw[bloc[it]] = pack2_bf16(dwp[o], dwp[o + CDIM]);
        }
        ka += TKK / 2;
        kb += (size_t)TKK * CDIM;
        __syncthreads();

        v16bf a = frag_A(As, lane, wm);
        v16bf b = frag_B(Bs, lane, wn);
        cacc = __builtin_amdgcn_wmma_f32_16x16x32_bf16(false, a, false, b,
                                                       (short)0, cacc, false, false);
        __syncthreads();
    }

    // tail slice: K = HMAIN..HDIM-1 (16 wide), guarded loads, zero padding
    {
#pragma unroll
        for (int it = 0; it < 2; ++it) {
            bool ok = (HMAIN + 2 * acp[it]) < HDIM;
            Aw[aloc[it]] = ok ? hbw[aoff[it] + ka] : 0u;
        }
#pragma unroll
        for (int it = 0; it < 4; ++it) {
            bool ok = (HMAIN + 2 * bkp[it]) < HDIM;
            float b0 = 0.0f, b1 = 0.0f;
            if (ok) { size_t o = boff[it] + kb; b0 = dwp[o]; b1 = dwp[o + CDIM]; }
            Bw[bloc[it]] = pack2_bf16(b0, b1);
        }
        __syncthreads();

        v16bf a = frag_A(As, lane, wm);
        v16bf b = frag_B(Bs, lane, wn);
        cacc = __builtin_amdgcn_wmma_f32_16x16x32_bf16(false, a, false, b,
                                                       (short)0, cacc, false, false);
        __syncthreads();
    }

    int mb = wm * 16 + ((lane < 16) ? 0 : 8);
    int nn = n0 + wn * 16 + (lane & 15);
#pragma unroll
    for (int v = 0; v < 8; ++v) {
        int m = mb + v;
        int tok = s_tok[m];
        if (tok >= 0)
            atomicAdd(&out[(size_t)tok * CDIM + nn], cacc[v]);
    }
}

// ---------------------------------------------------------------------------
// launch
// ---------------------------------------------------------------------------
extern "C" void kernel_launch(void* const* d_in, const int* in_sizes, int n_in,
                              void* d_out, int out_size, void* d_ws, size_t ws_size,
                              hipStream_t stream) {
    const float* x   = (const float*)d_in[0];
    const float* rw  = (const float*)d_in[1];
    const float* wg  = (const float*)d_in[2];
    const float* wu  = (const float*)d_in[3];
    const float* wd  = (const float*)d_in[4];
    const float* sg  = (const float*)d_in[5];
    const float* su  = (const float*)d_in[6];
    const float* sd  = (const float*)d_in[7];
    float* out = (float*)d_out;

    // workspace layout
    char* ws = (char*)d_ws;
    const size_t OFF_FILL = 0;                       // 16 * u32
    const size_t OFF_PSUM = 64;                      // 8 * f32
    const size_t OFF_LIST = 128;                     // 8*4096 * u32
    const size_t OFF_HW   = OFF_LIST + (size_t)EXP * TK_TOKENS * 4;
    size_t off_hbuf = OFF_HW + (size_t)TK_TOKENS * TOPK * 4;
    off_hbuf = (off_hbuf + 255) & ~(size_t)255;

    unsigned*       fill = (unsigned*)(ws + OFF_FILL);
    float*          psum = (float*)(ws + OFF_PSUM);
    unsigned*       list = (unsigned*)(ws + OFF_LIST);
    float*          hw   = (float*)(ws + OFF_HW);
    unsigned short* hbuf = (unsigned short*)(ws + off_hbuf);

    moe_zero_f32<<<4096, 256, 0, stream>>>(out, (size_t)TK_TOKENS * CDIM);
    moe_zero_ctr<<<1, 32, 0, stream>>>(fill, psum);

    moe_router<<<TK_TOKENS / 8, 256, 0, stream>>>(x, rw, fill, psum, list, hw);
    moe_aux<<<1, 32, 0, stream>>>(fill, psum, out);

    {
        dim3 grid((HDIM + 63) / 64, TK_TOKENS / 32, EXP + 1);
        moe_gateup<<<grid, 256, 0, stream>>>(x, wg, wu, sg, su,
                                             fill, list, hw, hbuf);
    }
    {
        dim3 grid(CDIM / 64, TK_TOKENS / 32, EXP + 1);
        moe_down<<<grid, 256, 0, stream>>>(hbuf, wd, sd, fill, list, out);
    }
}
